// KGAT_29901562314848
// MI455X (gfx1250) — matrix operand order repaired and checked by reading, not compiled, not run on hardware
//
#include <hip/hip_runtime.h>
#include <hip/hip_bf16.h>
#include <cstdint>

typedef __attribute__((ext_vector_type(16))) _Float16 v16h;
typedef __attribute__((ext_vector_type(2)))  _Float16 v2h;
typedef __attribute__((ext_vector_type(8)))  float    v8f;
typedef __attribute__((ext_vector_type(4)))  int      v4i;

#define N_ENT  50000
#define N_REL  16
#define D_E    64
#define D_R    64
#define D_OUT  32
#define N_EDGE 1000000

// ---------------- workspace layout ----------------
// proj (f16) : [N_REL, N_ENT, D_R] = 51,200,000 halves = 25.6M float slots
// then f32 arrays:
#define OFF_ATT   ((size_t)25600000)   // [N_EDGE]
#define OFF_EX    ((size_t)26600000)   // [N_EDGE]
#define OFF_NMAX  ((size_t)27600000)   // [N_ENT]
#define OFF_DEN   ((size_t)27650000)   // [N_ENT]
#define OFF_NH    ((size_t)27700000)   // [N_ENT, D_E]
// total ~ 30.9M floats = 123.6 MB

// gfx1250 async global->LDS path (guarded; falls back to plain loads)
#if defined(__has_builtin)
#  if __has_builtin(__builtin_amdgcn_global_load_async_to_lds_b128) && \
      __has_builtin(__builtin_amdgcn_s_wait_asynccnt)
#    define USE_ASYNC_LDS 1
#  endif
#endif

#if USE_ASYNC_LDS
// builtin signature (from hipcc diagnostic): v4i addrspace(1)*, v4i addrspace(3)*, imm, imm
typedef __attribute__((address_space(1))) v4i gv4i;
typedef __attribute__((address_space(3))) v4i lv4i;
#endif

// float atomic-max via signed/unsigned int monotonicity trick
// (valid given init to -inf)
__device__ __forceinline__ void atomicMaxF(float* addr, float val) {
    if (val >= 0.0f) {
        atomicMax((int*)addr, __float_as_int(val));
    } else {
        atomicMin((unsigned int*)addr, (unsigned int)__float_as_int(val));
    }
}

// ---------------- init: node_max=-inf, denom=0, N_h=0 ----------------
__global__ void kgat_init_kernel(float* node_max, float* denom, float* N_h) {
    int i = blockIdx.x * blockDim.x + threadIdx.x;
    if (i < N_ENT * D_E) N_h[i] = 0.0f;
    if (i < N_ENT) {
        node_max[i] = -__builtin_inff();
        denom[i]    = 0.0f;
    }
}

// ---------------- stage 1: proj[r] = entity_table @ W_R[r]  (f16 out) ------
// grid: (N_ENT/16, N_REL), block: 128 (4 waves). Wave w -> output cols [16w,16w+16).
// W_R[r] (64x64 f32, 16KB) staged to LDS via async global->LDS DMA.
__global__ void proj_gemm_kernel(const float* __restrict__ ent,
                                 const float* __restrict__ W_R,
                                 _Float16* __restrict__ proj) {
    __shared__ float Bs[D_E * D_R];               // 16 KB
    const int lane = threadIdx.x & 31;
    const int wave = threadIdx.x >> 5;            // 0..3
    const int row0 = blockIdx.x * 16;             // node tile base
    const int r    = blockIdx.y;                  // relation
    const int n0   = wave * 16;                   // D_R column tile
    const int m    = lane & 15;
    const int hi   = lane >> 4;                   // lane half-group

    const float* Bmat = W_R + (size_t)r * D_E * D_R;   // [k][n] row-major

#if USE_ASYNC_LDS
    {
        // 4096 floats = 1024 x b128 chunks; 128 threads x 8 chunks
        #pragma unroll
        for (int c = 0; c < 8; ++c) {
            const int idx = (threadIdx.x + c * 128) * 4;     // float index
            __builtin_amdgcn_global_load_async_to_lds_b128(
                (gv4i*)(uintptr_t)(Bmat + idx),
                (lv4i*)(uint32_t)(uintptr_t)(&Bs[idx]),
                0, 0);
        }
        __builtin_amdgcn_s_wait_asynccnt(0);
    }
    __syncthreads();
#else
    for (int idx = threadIdx.x; idx < D_E * D_R; idx += 128)
        Bs[idx] = Bmat[idx];
    __syncthreads();
#endif

    const float* Arow = ent + (size_t)(row0 + m) * D_E;

    v8f acc = {};
    #pragma unroll
    for (int s = 0; s < 2; ++s) {
        const int kb = s * 32;
        v16h a, b;
        #pragma unroll
        for (int i = 0; i < 8; ++i) {
            // A layout: lanes0-15 K0-7/K16-23 ; lanes16-31 K8-15/K24-31
            a[i]     = (_Float16)Arow[kb + hi * 8 + i];
            a[i + 8] = (_Float16)Arow[kb + 16 + hi * 8 + i];
        }
        #pragma unroll
        for (int j = 0; j < 16; ++j) {
            // B layout: lanes0-15 K0-7/K16-23 ; lanes16-31 K8-15/K24-31
            b[j] = (_Float16)Bs[(kb + (j < 8 ? hi * 8 + j : 16 + hi * 8 + (j - 8))) * D_R + (n0 + m)];
        }
        acc = __builtin_amdgcn_wmma_f32_16x16x32_f16(
            false, a, false, b, (short)0, acc, false, false);
    }
    _Float16* Out = proj + ((size_t)r * N_ENT + row0) * D_R;
    #pragma unroll
    for (int v = 0; v < 8; ++v) {
        Out[(size_t)(v + hi * 8) * D_R + (n0 + m)] = (_Float16)acc[v];
    }
}

// ---------------- stage 2: per-edge attention + segment max ----------------
// One wave per edge: each lane loads one packed half2 (4B) of each 128B row.
__global__ void edge_att_kernel(const _Float16* __restrict__ proj,
                                const float* __restrict__ rel,
                                const int* __restrict__ src,
                                const int* __restrict__ dst,
                                const int* __restrict__ etype,
                                float* __restrict__ att,
                                float* __restrict__ node_max) {
    const int e = blockIdx.x * (blockDim.x >> 5) + (threadIdx.x >> 5);
    if (e >= N_EDGE) return;
    const int lane = threadIdx.x & 31;
    const int r = etype[e], sn = src[e], dn = dst[e];
    const v2h hh = ((const v2h*)(proj + ((size_t)r * N_ENT + sn) * D_R))[lane];
    const v2h tt = ((const v2h*)(proj + ((size_t)r * N_ENT + dn) * D_R))[lane];
    const float2 rr = ((const float2*)(rel + (size_t)r * D_R))[lane];

    float sum = (float)hh[0] * tanhf((float)tt[0] + rr.x)
              + (float)hh[1] * tanhf((float)tt[1] + rr.y);
    #pragma unroll
    for (int off = 16; off > 0; off >>= 1)
        sum += __shfl_xor(sum, off, 32);
    if (lane == 0) {
        att[e] = sum;
        atomicMaxF(&node_max[dn], sum);
    }
}

// ---------------- stage 3: exp + segment sum ----------------
__global__ void edge_exp_kernel(const float* __restrict__ att,
                                const int* __restrict__ dst,
                                const float* __restrict__ node_max,
                                float* __restrict__ ex,
                                float* __restrict__ denom) {
    const int e = blockIdx.x * blockDim.x + threadIdx.x;
    if (e >= N_EDGE) return;
    const int dn = dst[e];
    const float v = __expf(att[e] - node_max[dn]);
    ex[e] = v;
    atomicAdd(&denom[dn], v);
}

// ---------------- stage 4: N_h[dst] += a * entity_table[src] ----------------
__global__ void edge_agg_kernel(const float* __restrict__ ent,
                                const int* __restrict__ src,
                                const int* __restrict__ dst,
                                const float* __restrict__ ex,
                                const float* __restrict__ denom,
                                float* __restrict__ N_h) {
    const int e = blockIdx.x * (blockDim.x >> 5) + (threadIdx.x >> 5);
    if (e >= N_EDGE) return;
    const int lane = threadIdx.x & 31;
    const int sn = src[e], dn = dst[e];
    const float a = ex[e] / denom[dn];
    const float2 xv = ((const float2*)(ent + (size_t)sn * D_E))[lane];
    float* h = N_h + (size_t)dn * D_E + 2 * lane;
    atomicAdd(&h[0], a * xv.x);
    atomicAdd(&h[1], a * xv.y);
}

// ---------------- stage 5: bi-interaction aggregator ----------------
// grid: N_ENT/16, block: 64 (2 waves). Wave w -> output cols [16w,16w+16).
// out = leaky((x+h)@W1 + b1) + leaky((x*h)@W2 + b2)
__global__ void biagg_kernel(const float* __restrict__ ent,
                             const float* __restrict__ N_h,
                             const float* __restrict__ W1,
                             const float* __restrict__ b1,
                             const float* __restrict__ W2,
                             const float* __restrict__ b2,
                             float* __restrict__ out) {
    const int lane = threadIdx.x & 31;
    const int wave = threadIdx.x >> 5;     // 0..1
    const int row0 = blockIdx.x * 16;
    const int n0   = wave * 16;
    const int m    = lane & 15;
    const int hi   = lane >> 4;

    const float* X = ent + (size_t)(row0 + m) * D_E;
    const float* H = N_h + (size_t)(row0 + m) * D_E;

    v8f acc1 = {}, acc2 = {};
    #pragma unroll
    for (int s = 0; s < 2; ++s) {
        const int kb = s * 32;
        v16h a1, a2, w1, w2;
        #pragma unroll
        for (int i = 0; i < 8; ++i) {
            int kA = kb + hi * 8 + i;
            int kB = kb + 16 + hi * 8 + i;
            float xa = X[kA], ha = H[kA];
            float xb = X[kB], hb = H[kB];
            a1[i]     = (_Float16)(xa + ha);
            a2[i]     = (_Float16)(xa * ha);
            a1[i + 8] = (_Float16)(xb + hb);
            a2[i + 8] = (_Float16)(xb * hb);
        }
        #pragma unroll
        for (int j = 0; j < 16; ++j) {
            int k = kb + (j < 8 ? hi * 8 + j : 16 + hi * 8 + (j - 8));
            w1[j] = (_Float16)W1[(size_t)k * D_OUT + (n0 + m)];
            w2[j] = (_Float16)W2[(size_t)k * D_OUT + (n0 + m)];
        }
        acc1 = __builtin_amdgcn_wmma_f32_16x16x32_f16(
            false, a1, false, w1, (short)0, acc1, false, false);
        acc2 = __builtin_amdgcn_wmma_f32_16x16x32_f16(
            false, a2, false, w2, (short)0, acc2, false, false);
    }
    const float bb1 = b1[n0 + m];
    const float bb2 = b2[n0 + m];
    #pragma unroll
    for (int v = 0; v < 8; ++v) {
        float o1 = acc1[v] + bb1;
        o1 = (o1 > 0.0f) ? o1 : 0.01f * o1;
        float o2 = acc2[v] + bb2;
        o2 = (o2 > 0.0f) ? o2 : 0.01f * o2;
        out[(size_t)(row0 + v + hi * 8) * D_OUT + (n0 + m)] = o1 + o2;
    }
}

extern "C" void kernel_launch(void* const* d_in, const int* in_sizes, int n_in,
                              void* d_out, int out_size, void* d_ws, size_t ws_size,
                              hipStream_t stream) {
    const float* ent   = (const float*)d_in[0];  // [N_ENT, D_E]
    const float* rel   = (const float*)d_in[1];  // [N_REL, D_R]
    const float* W_R   = (const float*)d_in[2];  // [N_REL, D_E, D_R]
    const float* W1    = (const float*)d_in[3];  // [D_E, D_OUT]
    const float* b1    = (const float*)d_in[4];  // [D_OUT]
    const float* W2    = (const float*)d_in[5];  // [D_E, D_OUT]
    const float* b2    = (const float*)d_in[6];  // [D_OUT]
    const int*   src   = (const int*)d_in[7];    // [N_EDGE]
    const int*   dst   = (const int*)d_in[8];    // [N_EDGE]
    const int*   etype = (const int*)d_in[9];    // [N_EDGE]
    float* out = (float*)d_out;

    float*    ws       = (float*)d_ws;
    _Float16* proj     = (_Float16*)d_ws;        // 102.4 MB, L2-resident
    float*    att      = ws + OFF_ATT;
    float*    ex       = ws + OFF_EX;
    float*    node_max = ws + OFF_NMAX;
    float*    denom    = ws + OFF_DEN;
    float*    N_h      = ws + OFF_NH;

    // init segment buffers
    {
        int n = N_ENT * D_E;
        kgat_init_kernel<<<(n + 255) / 256, 256, 0, stream>>>(node_max, denom, N_h);
    }
    // stage 1: per-relation projection (WMMA, async LDS staging of W_R)
    {
        dim3 grid(N_ENT / 16, N_REL);
        proj_gemm_kernel<<<grid, 128, 0, stream>>>(ent, W_R, proj);
    }
    // stage 2: edge attention + segment max (wave per edge, f16 gathers)
    edge_att_kernel<<<N_EDGE / 8, 256, 0, stream>>>(proj, rel, src, dst, etype,
                                                    att, node_max);
    // stage 3: exp + segment sum
    edge_exp_kernel<<<(N_EDGE + 255) / 256, 256, 0, stream>>>(att, dst, node_max,
                                                              ex, denom);
    // stage 4: weighted message aggregation (wave per edge)
    edge_agg_kernel<<<N_EDGE / 8, 256, 0, stream>>>(ent, src, dst, ex, denom, N_h);
    // stage 5: bi-interaction aggregator (WMMA) -> d_out
    biagg_kernel<<<N_ENT / 16, 64, 0, stream>>>(ent, N_h, W1, b1, W2, b2, out);
}
